// SAGEEncoder_50955492000382
// MI455X (gfx1250) — matrix operand order repaired and checked by reading, not compiled, not run on hardware
//
#include <hip/hip_runtime.h>

// ---------------------------------------------------------------------------
// GraphSAGE 2-layer encoder for MI455X (gfx1250, wave32, WMMA)
//   layer: out = [h | mean_agg(h)] @ [W_self ; W_neigh] + b
// Aggregation: edge-parallel float4 gather + f32 atomic scatter (hits L2).
// GEMM: v_wmma_f32_16x16x32_f16, one wave per 16x16 tile, K=256 fused.
// ---------------------------------------------------------------------------

typedef _Float16 half4  __attribute__((ext_vector_type(4)));
typedef _Float16 half8  __attribute__((ext_vector_type(8)));
typedef _Float16 half16 __attribute__((ext_vector_type(16)));
typedef float    float8 __attribute__((ext_vector_type(8)));

static constexpr int   NNODES = 50000;
static constexpr int   NEDGES = 800000;
static constexpr int   FIN    = 128;   // IN == HID == 128
static constexpr int   FOUT   = 64;
static constexpr int   KF     = 256;   // fused K = 128 (self) + 128 (neigh)
static constexpr int   MTILES = NNODES / 16;  // 3125, exact

// ---------------------------------------------------------------- utilities
__global__ void zero_f32(float* __restrict__ p, long n) {
  long i = blockIdx.x * (long)blockDim.x + threadIdx.x;
  long stride = (long)gridDim.x * blockDim.x;
  for (; i < n; i += stride) p[i] = 0.0f;
}

// Bt[n][k] = (k<128 ? Wself[k][n] : Wneigh[k-128][n]) as f16; Bt is [Ncols][256]
__global__ void pack_weights(const float* __restrict__ Wself,
                             const float* __restrict__ Wneigh,
                             _Float16* __restrict__ Bt, int Ncols) {
  int idx = blockIdx.x * blockDim.x + threadIdx.x;
  if (idx >= Ncols * KF) return;
  int nn = idx >> 8;          // / 256
  int k  = idx & (KF - 1);
  float v = (k < FIN) ? Wself[k * Ncols + nn] : Wneigh[(k - FIN) * Ncols + nn];
  Bt[(long)nn * KF + k] = (_Float16)v;
}

// ------------------------------------------------------------- aggregation
__global__ void degree_kernel(const int* __restrict__ dst, float* __restrict__ deg) {
  int e = blockIdx.x * blockDim.x + threadIdx.x;
  if (e < NEDGES) atomicAdd(&deg[dst[e]], 1.0f);
}

// one thread = (edge, 4-float chunk); 32 chunks of 4 per edge
__global__ void scatter_f32(const float* __restrict__ feat,
                            const int* __restrict__ src,
                            const int* __restrict__ dst,
                            float* __restrict__ aggsum) {
  long idx = blockIdx.x * (long)blockDim.x + threadIdx.x;
  if (idx >= (long)NEDGES * 32) return;
  int e  = (int)(idx >> 5);
  int ch = ((int)idx & 31) * 4;
  int s = src[e], d = dst[e];
  const float4 v = *(const float4*)(feat + (long)s * FIN + ch);
  float* o = aggsum + (long)d * FIN + ch;
  atomicAdd(o + 0, v.x);
  atomicAdd(o + 1, v.y);
  atomicAdd(o + 2, v.z);
  atomicAdd(o + 3, v.w);
}

// gather f16 hidden state (stride KF, cols 0..127), scatter f32 sums
__global__ void scatter_f16(const _Float16* __restrict__ h,
                            const int* __restrict__ src,
                            const int* __restrict__ dst,
                            float* __restrict__ aggsum) {
  long idx = blockIdx.x * (long)blockDim.x + threadIdx.x;
  if (idx >= (long)NEDGES * 32) return;
  int e  = (int)(idx >> 5);
  int ch = ((int)idx & 31) * 4;
  int s = src[e], d = dst[e];
  half4 v = *(const half4*)(h + (long)s * KF + ch);
  float* o = aggsum + (long)d * FIN + ch;
  atomicAdd(o + 0, (float)v[0]);
  atomicAdd(o + 1, (float)v[1]);
  atomicAdd(o + 2, (float)v[2]);
  atomicAdd(o + 3, (float)v[3]);
}

// A1[n][0:128] = f16(feat[n]); A1[n][128:256] = f16(aggsum[n]/max(deg,1))
__global__ void finalize1(const float* __restrict__ feat,
                          const float* __restrict__ aggsum,
                          const float* __restrict__ deg,
                          _Float16* __restrict__ A1) {
  long idx = blockIdx.x * (long)blockDim.x + threadIdx.x;
  if (idx >= (long)NNODES * FIN) return;
  int n = (int)(idx >> 7);
  int f = (int)idx & (FIN - 1);
  float dv = fmaxf(deg[n], 1.0f);
  A1[(long)n * KF + f]       = (_Float16)feat[idx];
  A1[(long)n * KF + FIN + f] = (_Float16)(aggsum[idx] / dv);
}

// A2[n][128:256] = f16(aggsum[n]/max(deg,1))   (first half written by GEMM1)
__global__ void finalize2(const float* __restrict__ aggsum,
                          const float* __restrict__ deg,
                          _Float16* __restrict__ A2) {
  long idx = blockIdx.x * (long)blockDim.x + threadIdx.x;
  if (idx >= (long)NNODES * FIN) return;
  int n = (int)(idx >> 7);
  int f = (int)idx & (FIN - 1);
  float dv = fmaxf(deg[n], 1.0f);
  A2[(long)n * KF + FIN + f] = (_Float16)(aggsum[idx] / dv);
}

// ------------------------------------------------------------------- GEMM
// C[M][N] = A[M][KF] * Bt[N][KF]^T + bias, optional relu.
// One wave -> one 16x16 tile; blockDim.x/32 waves tile N; blockIdx.x tiles M.
// Fragment layouts per CDNA5 ISA 7.12.2:
//   A 16x32 f16 : lane g=lane>>4, row=lane&15; halfs[0:8]=K(k0+8g..), halfs[8:16]=K(k0+16+8g..)
//   B 32x16 f16 : col=lane&15; halfs[0:8]=K(k0+8g..), halfs[8:16]=K(k0+16+8g..) from Bt[col][k]
//   C/D 16x16 f32: acc[r] -> (row = m0 + r + 8g, col = n0 + (lane&15))
__global__ void sage_gemm_wmma(const _Float16* __restrict__ A,
                               const _Float16* __restrict__ Bt,
                               const float* __restrict__ bias,
                               float* __restrict__ outF32,      // used if outF16 == null
                               _Float16* __restrict__ outF16,   // f16 out, stride ld16
                               int N, int ld16, int relu) {
  const int lane = threadIdx.x & 31;
  const int wave = threadIdx.x >> 5;
  const int g    = lane >> 4;
  const int lr   = lane & 15;
  const int m0   = blockIdx.x * 16;
  const int n0   = wave * 16;

  const _Float16* arow = A  + (long)(m0 + lr) * KF;
  const _Float16* bcol = Bt + (long)(n0 + lr) * KF;

  float8 acc = {};
#pragma unroll
  for (int kk = 0; kk < KF / 32; ++kk) {
    const int k0 = kk * 32;
    half8 alo = *(const half8*)(arow + k0 + 8 * g);
    half8 ahi = *(const half8*)(arow + k0 + 16 + 8 * g);
    half16 a = __builtin_shufflevector(alo, ahi, 0, 1, 2, 3, 4, 5, 6, 7,
                                       8, 9, 10, 11, 12, 13, 14, 15);
    half8 blo = *(const half8*)(bcol + k0 + 8 * g);
    half8 bhi = *(const half8*)(bcol + k0 + 16 + 8 * g);
    half16 b = __builtin_shufflevector(blo, bhi, 0, 1, 2, 3, 4, 5, 6, 7,
                                       8, 9, 10, 11, 12, 13, 14, 15);
    acc = __builtin_amdgcn_wmma_f32_16x16x32_f16(
        /*neg_a=*/false, a, /*neg_b=*/false, b,
        /*c_mod=*/(short)0, acc, /*reuse_a=*/false, /*reuse_b=*/false);
  }

  const int col = n0 + lr;
  const float bv = bias[col];
#pragma unroll
  for (int r = 0; r < 8; ++r) {
    const int row = m0 + r + 8 * g;
    float v = acc[r] + bv;
    if (relu) v = fmaxf(v, 0.0f);
    if (outF16) outF16[(long)row * ld16 + col] = (_Float16)v;
    else        outF32[(long)row * N + col] = v;
  }
}

// ------------------------------------------------------------------ launch
extern "C" void kernel_launch(void* const* d_in, const int* in_sizes, int n_in,
                              void* d_out, int out_size, void* d_ws, size_t ws_size,
                              hipStream_t stream) {
  (void)in_sizes; (void)n_in; (void)out_size; (void)ws_size;

  const float* feat    = (const float*)d_in[0];
  const int*   src     = (const int*)d_in[1];
  const int*   dst     = (const int*)d_in[2];
  const float* Wself1  = (const float*)d_in[3];
  const float* Wneigh1 = (const float*)d_in[4];
  const float* b1      = (const float*)d_in[5];
  const float* Wself2  = (const float*)d_in[6];
  const float* Wneigh2 = (const float*)d_in[7];
  const float* b2      = (const float*)d_in[8];
  float* out = (float*)d_out;

  // workspace layout (256-byte aligned slabs)
  char* ws = (char*)d_ws;
  size_t off = 0;
  auto take = [&](size_t bytes) {
    char* p = ws + off;
    off += (bytes + 255) & ~(size_t)255;
    return p;
  };
  float*     aggsum = (float*)take((size_t)NNODES * FIN * sizeof(float));   // 25.6 MB
  float*     deg    = (float*)take((size_t)NNODES * sizeof(float));         // 0.2 MB
  _Float16*  A1     = (_Float16*)take((size_t)NNODES * KF * sizeof(_Float16)); // 25.6 MB
  _Float16*  A2     = (_Float16*)take((size_t)NNODES * KF * sizeof(_Float16)); // 25.6 MB
  _Float16*  Bt1    = (_Float16*)take((size_t)FIN  * KF * sizeof(_Float16));   // 64 KB
  _Float16*  Bt2    = (_Float16*)take((size_t)FOUT * KF * sizeof(_Float16));   // 32 KB

  const long aggN   = (long)NNODES * FIN;
  const long edgeN  = (long)NEDGES * 32;
  const int  T      = 256;
  const int  edgeGrid = (int)((edgeN + T - 1) / T);
  const int  nodeGrid = (int)((aggN + T - 1) / T);

  // ---- prep: clear accumulators, pack transposed f16 weights, degrees ----
  zero_f32<<<4096, T, 0, stream>>>(aggsum, aggN);
  zero_f32<<<256, T, 0, stream>>>(deg, NNODES);
  pack_weights<<<(FIN * KF + T - 1) / T, T, 0, stream>>>(Wself1, Wneigh1, Bt1, FIN);
  pack_weights<<<(FOUT * KF + T - 1) / T, T, 0, stream>>>(Wself2, Wneigh2, Bt2, FOUT);
  degree_kernel<<<(NEDGES + T - 1) / T, T, 0, stream>>>(dst, deg);

  // ---- layer 1 aggregation + fused GEMM (relu, f16 out into A2[:, :128]) ----
  scatter_f32<<<edgeGrid, T, 0, stream>>>(feat, src, dst, aggsum);
  finalize1<<<nodeGrid, T, 0, stream>>>(feat, aggsum, deg, A1);
  sage_gemm_wmma<<<MTILES, (FIN / 16) * 32, 0, stream>>>(
      A1, Bt1, b1, nullptr, A2, FIN, KF, /*relu=*/1);

  // ---- layer 2 aggregation + fused GEMM (f32 out) ----
  zero_f32<<<4096, T, 0, stream>>>(aggsum, aggN);
  scatter_f16<<<edgeGrid, T, 0, stream>>>(A2, src, dst, aggsum);
  finalize2<<<nodeGrid, T, 0, stream>>>(aggsum, deg, A2);
  sage_gemm_wmma<<<MTILES, (FOUT / 16) * 32, 0, stream>>>(
      A2, Bt2, b2, out, nullptr, FOUT, 0, /*relu=*/0);
}
